// LlamaMoEMLP_84267258347975
// MI455X (gfx1250) — compile-verified, matrix operand order checked by reading, not run on hardware
//
#include <hip/hip_runtime.h>
#include <hip/hip_bf16.h>
#include <math.h>

#define TOK 4096
#define HD  2048
#define ID  5504
#define NE  8

typedef __attribute__((ext_vector_type(16))) __bf16 v16bf;
typedef __attribute__((ext_vector_type(8)))  float  v8f;

__device__ __forceinline__ unsigned short f2bf(float f) {
  unsigned int u = __float_as_uint(f);
  u += 0x7FFFu + ((u >> 16) & 1u);   // round-to-nearest-even
  return (unsigned short)(u >> 16);
}

// Packed fp32x2 -> bf16x2 (v_cvt_pk_bf16_f32 on gfx1250)
__device__ __forceinline__ unsigned int pk_bf16(float lo, float hi) {
  union { __hip_bfloat162 h2; unsigned int u; } cv;
  cv.h2 = __float22bfloat162_rn(make_float2(lo, hi));
  return cv.u;
}

union Frag { v16bf v; unsigned int u[8]; };

// Load a 16x32 bf16 WMMA fragment from an LDS tile laid out [row][k] with
// 'stride' ushorts per row. Per ISA 7.12.2: lanes 0-15 -> K {0..7,16..23},
// lanes 16-31 -> K {8..15,24..31}; row = lane&15 for both A (M) and B (N).
__device__ __forceinline__ v16bf ld_frag(const unsigned short* base, int row,
                                         int stride, int kbase, int half) {
  Frag f;
  const unsigned short* rp = base + row * stride + kbase + half * 8;
#pragma unroll
  for (int i = 0; i < 8; ++i) {
    int k = ((i & 4) << 2) + ((i & 3) << 1);  // (i/4)*16 + (i%4)*2
    f.u[i] = *(const unsigned int*)(rp + k);  // two consecutive bf16 (K, K+1)
  }
  return f.v;
}

// ---------------- Kernel 1: convert x -> bf16, zero out, zero counters ----
__global__ void prep_kernel(const float* __restrict__ x,
                            unsigned short* __restrict__ xb,
                            float* __restrict__ out, int* __restrict__ counts) {
  size_t n = (size_t)TOK * HD;
  for (size_t i = (size_t)blockIdx.x * blockDim.x + threadIdx.x; i < n;
       i += (size_t)gridDim.x * blockDim.x) {
    xb[i] = f2bf(x[i]);
    out[i] = 0.f;
  }
  if (blockIdx.x == 0 && threadIdx.x < NE) counts[threadIdx.x] = 0;
}

// ---------------- Kernel 2: router (one wave32 per token) -----------------
__global__ void router_kernel(const float* __restrict__ x,
                              const float* __restrict__ rw,
                              int* __restrict__ top_idx,
                              float* __restrict__ top_w,
                              int* __restrict__ counts) {
  int gtid = blockIdx.x * blockDim.x + threadIdx.x;
  int t = gtid >> 5;
  int lane = gtid & 31;
  if (t >= TOK) return;
  float acc[NE];
#pragma unroll
  for (int e = 0; e < NE; ++e) acc[e] = 0.f;
  const float* xr = x + (size_t)t * HD;
  for (int k = lane; k < HD; k += 32) {
    float xv = xr[k];
#pragma unroll
    for (int e = 0; e < NE; ++e) acc[e] += xv * rw[e * HD + k];
  }
#pragma unroll
  for (int e = 0; e < NE; ++e) {
#pragma unroll
    for (int m = 16; m > 0; m >>= 1) acc[e] += __shfl_xor(acc[e], m, 32);
  }
  if (lane == 0) {
    int e0 = 0;
#pragma unroll
    for (int e = 1; e < NE; ++e) if (acc[e] > acc[e0]) e0 = e;
    int e1 = (e0 == 0) ? 1 : 0;
#pragma unroll
    for (int e = 0; e < NE; ++e) if (e != e0 && acc[e] > acc[e1]) e1 = e;
    float w1 = 1.f / (1.f + expf(acc[e0] - acc[e1]));  // softmax over top-2
    float w0 = 1.f - w1;
    top_idx[t * 2]     = e0;  top_idx[t * 2 + 1] = e1;
    top_w[t * 2]       = w0;  top_w[t * 2 + 1]   = w1;
    atomicAdd(&counts[e0], 1);
    atomicAdd(&counts[e1], 1);
  }
}

// ---------------- Kernel 3: tiny exclusive scan over 8 counts -------------
__global__ void scan_kernel(const int* __restrict__ counts,
                            int* __restrict__ offsets, int* __restrict__ cursor) {
  if (blockIdx.x == 0 && threadIdx.x == 0) {
    int s = 0;
    for (int e = 0; e < NE; ++e) { offsets[e] = s; cursor[e] = s; s += counts[e]; }
    offsets[NE] = s;
  }
}

// ---------------- Kernel 4: build per-expert assignment lists -------------
__global__ void fill_kernel(const int* __restrict__ top_idx,
                            const float* __restrict__ top_w,
                            int* __restrict__ cursor,
                            int* __restrict__ atok, float* __restrict__ aw) {
  int t = blockIdx.x * blockDim.x + threadIdx.x;
  if (t >= TOK) return;
#pragma unroll
  for (int k = 0; k < 2; ++k) {
    int e = top_idx[t * 2 + k];
    int slot = atomicAdd(&cursor[e], 1);
    atok[slot] = t;
    aw[slot] = top_w[t * 2 + k];
  }
}

// ---------------- Kernel 5: gate/up GEMMs + silu*mul -> hbuf (bf16) -------
// Block tile: M=64 (gathered rows) x N=128 (I cols). 8 waves in 2x4 grid,
// each wave owns a 32x32 output tile => 8 WMMAs per 32-wide k-step from
// 6 LDS fragment loads. Next k-chunk is prefetched into registers (with
// packed bf16 convert) while the current chunk's WMMAs run.
__global__ __launch_bounds__(256) void gateup_kernel(
    const unsigned short* __restrict__ xb,
    const float* __restrict__ Wg, const float* __restrict__ Wu,
    const int* __restrict__ offsets, const int* __restrict__ atok,
    unsigned short* __restrict__ hbuf) {
  __shared__ __align__(16) unsigned short As[64][66];
  __shared__ __align__(16) unsigned short Gs[128][66];
  __shared__ __align__(16) unsigned short Us[128][66];
  __shared__ int tokS[64];

  const int e = blockIdx.z;
  const int off = offsets[e];
  const int cnt = offsets[e + 1] - off;
  const int tilebase = blockIdx.x * 64;
  if (tilebase >= cnt) return;                 // uniform early-exit
  const int colbase = blockIdx.y * 128;        // over I (5504 = 128*43)
  const int tid = threadIdx.x;

  if (tid < 64) {
    int idx = tilebase + tid;
    tokS[tid] = (idx < cnt) ? atok[off + idx] : 0;
  }
  __syncthreads();

  const int lane = tid & 31, wave = tid >> 5;
  const int m0 = (wave & 1) * 32;              // 2 wave-rows cover M=64
  const int n0 = (wave >> 1) * 32;             // 4 wave-cols cover N=128
  const int half = lane >> 4, lrow = lane & 15;

  v8f accG[2][2] = {{{}, {}}, {{}, {}}};
  v8f accU[2][2] = {{{}, {}}, {{}, {}}};
  const size_t wbase = ((size_t)e * ID + colbase) * HD;

  // Register prefetch buffers (bf16-packed)
  uint2 pa[4];
  unsigned int pg[8][2], pu[8][2];

  auto load_chunk = [&](int k0) {
#pragma unroll
    for (int i = 0; i < 4; ++i) {
      int q = i * 256 + tid;                   // quad of 4 bf16
      int r = q >> 4, c4 = (q & 15) * 4;
      pa[i] = make_uint2(0u, 0u);
      if (tilebase + r < cnt)
        pa[i] = *(const uint2*)(xb + (size_t)tokS[r] * HD + k0 + c4);
    }
#pragma unroll
    for (int i = 0; i < 8; ++i) {
      int q = i * 256 + tid;
      int r = q >> 4, c4 = (q & 15) * 4;
      size_t goff = wbase + (size_t)r * HD + k0 + c4;
      float4 g = *(const float4*)(Wg + goff);
      float4 u = *(const float4*)(Wu + goff);
      pg[i][0] = pk_bf16(g.x, g.y);
      pg[i][1] = pk_bf16(g.z, g.w);
      pu[i][0] = pk_bf16(u.x, u.y);
      pu[i][1] = pk_bf16(u.z, u.w);
    }
  };
  auto store_chunk = [&]() {
#pragma unroll
    for (int i = 0; i < 4; ++i) {
      int q = i * 256 + tid;
      int r = q >> 4, c4 = (q & 15) * 4;
      *(unsigned int*)&As[r][c4]     = pa[i].x;
      *(unsigned int*)&As[r][c4 + 2] = pa[i].y;
    }
#pragma unroll
    for (int i = 0; i < 8; ++i) {
      int q = i * 256 + tid;
      int r = q >> 4, c4 = (q & 15) * 4;
      *(unsigned int*)&Gs[r][c4]     = pg[i][0];
      *(unsigned int*)&Gs[r][c4 + 2] = pg[i][1];
      *(unsigned int*)&Us[r][c4]     = pu[i][0];
      *(unsigned int*)&Us[r][c4 + 2] = pu[i][1];
    }
  };

  load_chunk(0);
  for (int k0 = 0; k0 < HD; k0 += 64) {
    store_chunk();
    __syncthreads();
    if (k0 + 64 < HD) load_chunk(k0 + 64);     // HBM loads fly under WMMAs
#pragma unroll
    for (int ks = 0; ks < 64; ks += 32) {
      v16bf a0 = ld_frag(&As[0][0], m0 + lrow, 66, ks, half);
      v16bf a1 = ld_frag(&As[0][0], m0 + 16 + lrow, 66, ks, half);
      v16bf g0 = ld_frag(&Gs[0][0], n0 + lrow, 66, ks, half);
      v16bf g1 = ld_frag(&Gs[0][0], n0 + 16 + lrow, 66, ks, half);
      v16bf u0 = ld_frag(&Us[0][0], n0 + lrow, 66, ks, half);
      v16bf u1 = ld_frag(&Us[0][0], n0 + 16 + lrow, 66, ks, half);
      accG[0][0] = __builtin_amdgcn_wmma_f32_16x16x32_bf16(false, a0, false, g0, (short)0, accG[0][0], false, false);
      accG[0][1] = __builtin_amdgcn_wmma_f32_16x16x32_bf16(false, a0, false, g1, (short)0, accG[0][1], false, false);
      accG[1][0] = __builtin_amdgcn_wmma_f32_16x16x32_bf16(false, a1, false, g0, (short)0, accG[1][0], false, false);
      accG[1][1] = __builtin_amdgcn_wmma_f32_16x16x32_bf16(false, a1, false, g1, (short)0, accG[1][1], false, false);
      accU[0][0] = __builtin_amdgcn_wmma_f32_16x16x32_bf16(false, a0, false, u0, (short)0, accU[0][0], false, false);
      accU[0][1] = __builtin_amdgcn_wmma_f32_16x16x32_bf16(false, a0, false, u1, (short)0, accU[0][1], false, false);
      accU[1][0] = __builtin_amdgcn_wmma_f32_16x16x32_bf16(false, a1, false, u0, (short)0, accU[1][0], false, false);
      accU[1][1] = __builtin_amdgcn_wmma_f32_16x16x32_bf16(false, a1, false, u1, (short)0, accU[1][1], false, false);
    }
    __syncthreads();
  }

#pragma unroll
  for (int mi = 0; mi < 2; ++mi) {
#pragma unroll
    for (int v = 0; v < 8; ++v) {
      int rloc = m0 + mi * 16 + v + half * 8;  // C/D layout: M = v + 8*half
      if (tilebase + rloc < cnt) {
        size_t rowoff = (size_t)(off + tilebase + rloc) * ID;
#pragma unroll
        for (int ni = 0; ni < 2; ++ni) {
          float g = accG[mi][ni][v], u = accU[mi][ni][v];
          float hv = (g / (1.f + expf(-g))) * u;
          hbuf[rowoff + colbase + n0 + ni * 16 + lrow] = f2bf(hv);
        }
      }
    }
  }
}

// ---------------- Kernel 6: down GEMM, scale by route weight, scatter -----
// Block tile: M=64 x N=128 (H cols, 2048 = 128*16); wave tile 32x32.
__global__ __launch_bounds__(256) void down_kernel(
    const unsigned short* __restrict__ hbuf, const float* __restrict__ Wd,
    const int* __restrict__ offsets, const int* __restrict__ atok,
    const float* __restrict__ aw, float* __restrict__ out) {
  __shared__ __align__(16) unsigned short As[64][66];
  __shared__ __align__(16) unsigned short Ds[128][66];
  __shared__ int tokS[64];
  __shared__ float wS[64];

  const int e = blockIdx.z;
  const int off = offsets[e];
  const int cnt = offsets[e + 1] - off;
  const int tilebase = blockIdx.x * 64;
  if (tilebase >= cnt) return;
  const int colbase = blockIdx.y * 128;        // over H
  const int tid = threadIdx.x;

  if (tid < 64) {
    int idx = tilebase + tid;
    tokS[tid] = (idx < cnt) ? atok[off + idx] : 0;
    wS[tid]   = (idx < cnt) ? aw[off + idx] : 0.f;
  }
  __syncthreads();

  const int lane = tid & 31, wave = tid >> 5;
  const int m0 = (wave & 1) * 32, n0 = (wave >> 1) * 32;
  const int half = lane >> 4, lrow = lane & 15;

  v8f acc[2][2] = {{{}, {}}, {{}, {}}};
  const size_t wdbase = ((size_t)e * HD + colbase) * ID;

  uint2 pa[4];
  unsigned int pd[8][2];

  auto load_chunk = [&](int k0) {
#pragma unroll
    for (int i = 0; i < 4; ++i) {
      int q = i * 256 + tid;
      int r = q >> 4, c4 = (q & 15) * 4;
      pa[i] = make_uint2(0u, 0u);
      if (tilebase + r < cnt)
        pa[i] = *(const uint2*)(hbuf + (size_t)(off + tilebase + r) * ID + k0 + c4);
    }
#pragma unroll
    for (int i = 0; i < 8; ++i) {
      int q = i * 256 + tid;
      int r = q >> 4, c4 = (q & 15) * 4;
      float4 d = *(const float4*)(Wd + wdbase + (size_t)r * ID + k0 + c4);
      pd[i][0] = pk_bf16(d.x, d.y);
      pd[i][1] = pk_bf16(d.z, d.w);
    }
  };
  auto store_chunk = [&]() {
#pragma unroll
    for (int i = 0; i < 4; ++i) {
      int q = i * 256 + tid;
      int r = q >> 4, c4 = (q & 15) * 4;
      *(unsigned int*)&As[r][c4]     = pa[i].x;
      *(unsigned int*)&As[r][c4 + 2] = pa[i].y;
    }
#pragma unroll
    for (int i = 0; i < 8; ++i) {
      int q = i * 256 + tid;
      int r = q >> 4, c4 = (q & 15) * 4;
      *(unsigned int*)&Ds[r][c4]     = pd[i][0];
      *(unsigned int*)&Ds[r][c4 + 2] = pd[i][1];
    }
  };

  load_chunk(0);
  for (int k0 = 0; k0 < ID; k0 += 64) {        // ID = 64*86 exactly
    store_chunk();
    __syncthreads();
    if (k0 + 64 < ID) load_chunk(k0 + 64);
#pragma unroll
    for (int ks = 0; ks < 64; ks += 32) {
      v16bf a0 = ld_frag(&As[0][0], m0 + lrow, 66, ks, half);
      v16bf a1 = ld_frag(&As[0][0], m0 + 16 + lrow, 66, ks, half);
      v16bf b0 = ld_frag(&Ds[0][0], n0 + lrow, 66, ks, half);
      v16bf b1 = ld_frag(&Ds[0][0], n0 + 16 + lrow, 66, ks, half);
      acc[0][0] = __builtin_amdgcn_wmma_f32_16x16x32_bf16(false, a0, false, b0, (short)0, acc[0][0], false, false);
      acc[0][1] = __builtin_amdgcn_wmma_f32_16x16x32_bf16(false, a0, false, b1, (short)0, acc[0][1], false, false);
      acc[1][0] = __builtin_amdgcn_wmma_f32_16x16x32_bf16(false, a1, false, b0, (short)0, acc[1][0], false, false);
      acc[1][1] = __builtin_amdgcn_wmma_f32_16x16x32_bf16(false, a1, false, b1, (short)0, acc[1][1], false, false);
    }
    __syncthreads();
  }

#pragma unroll
  for (int mi = 0; mi < 2; ++mi) {
#pragma unroll
    for (int v = 0; v < 8; ++v) {
      int rloc = m0 + mi * 16 + v + half * 8;
      if (tilebase + rloc < cnt) {
        int t = tokS[rloc];
        float w = wS[rloc];
#pragma unroll
        for (int ni = 0; ni < 2; ++ni) {
          atomicAdd(&out[(size_t)t * HD + colbase + n0 + ni * 16 + lrow],
                    acc[mi][ni][v] * w);
        }
      }
    }
  }
}

extern "C" void kernel_launch(void* const* d_in, const int* in_sizes, int n_in,
                              void* d_out, int out_size, void* d_ws, size_t ws_size,
                              hipStream_t stream) {
  (void)in_sizes; (void)n_in; (void)out_size; (void)ws_size;
  const float* x  = (const float*)d_in[0];
  const float* rw = (const float*)d_in[1];
  const float* Wg = (const float*)d_in[2];
  const float* Wu = (const float*)d_in[3];
  const float* Wd = (const float*)d_in[4];
  float* out = (float*)d_out;

  char* ws = (char*)d_ws;
  size_t o = 0;
  auto take = [&](size_t bytes) -> char* {
    char* p = ws + o;
    o = (o + bytes + 255) & ~(size_t)255;
    return p;
  };
  int*            counts  = (int*)take(NE * sizeof(int));
  int*            offsets = (int*)take((NE + 1) * sizeof(int));
  int*            cursor  = (int*)take(NE * sizeof(int));
  int*            top_idx = (int*)take((size_t)TOK * 2 * sizeof(int));
  float*          top_w   = (float*)take((size_t)TOK * 2 * sizeof(float));
  int*            atok    = (int*)take((size_t)TOK * 2 * sizeof(int));
  float*          aw      = (float*)take((size_t)TOK * 2 * sizeof(float));
  unsigned short* xb      = (unsigned short*)take((size_t)TOK * HD * sizeof(unsigned short));
  unsigned short* hbuf    = (unsigned short*)take((size_t)TOK * 2 * ID * sizeof(unsigned short));

  prep_kernel<<<2048, 256, 0, stream>>>(x, xb, out, counts);
  router_kernel<<<(TOK * 32) / 256, 256, 0, stream>>>(x, rw, top_idx, top_w, counts);
  scan_kernel<<<1, 1, 0, stream>>>(counts, offsets, cursor);
  fill_kernel<<<TOK / 256, 256, 0, stream>>>(top_idx, top_w, cursor, atok, aw);

  dim3 gridC(TOK / 64, ID / 128, NE);  // worst-case M tiles; empty tiles exit
  gateup_kernel<<<gridC, 256, 0, stream>>>(xb, Wg, Wu, offsets, atok, hbuf);
  dim3 gridD(TOK / 64, HD / 128, NE);
  down_kernel<<<gridD, 256, 0, stream>>>(hbuf, Wd, offsets, atok, aw, out);
}
